// FlashTalkAudioCrossAttention_29652454212083
// MI455X (gfx1250) — matrix operand order repaired and checked by reading, not compiled, not run on hardware
//
#include <hip/hip_runtime.h>
#include <hip/hip_bf16.h>

// ---------------------------------------------------------------------------
// Types for CDNA5 WMMA (wave32): A/B fragments are 16 bf16 per lane (8 VGPRs),
// C/D accumulator is 8 f32 per lane (8 VGPRs).
// ---------------------------------------------------------------------------
typedef __attribute__((ext_vector_type(16))) __bf16 v16bf;
typedef __attribute__((ext_vector_type(8)))  float  v8f;
typedef __attribute__((ext_vector_type(4)))  unsigned int v4u;

union Frag {
    v16bf v;
    v4u   q[2];           // two 16B chunks
    unsigned short u[16];
};

#define WMMA_BF16(A, B, C) \
    __builtin_amdgcn_wmma_f32_16x16x32_bf16(false, (A), false, (B), (short)0, (C), false, false)

// round-to-nearest-even f32 -> bf16 bits
__device__ __forceinline__ unsigned short f2bf(float x) {
    unsigned int u = __float_as_uint(x);
    u += 0x7FFFu + ((u >> 16) & 1u);
    return (unsigned short)(u >> 16);
}

// ---------------------------------------------------------------------------
// Problem constants (setup_inputs: B=1, N_t=21, S=1560, C=2048, H=32, D=64,
// N_a=32, A=768).
// ---------------------------------------------------------------------------
#define NTOK   32760          // N_t * S
#define CDIM   2048
#define HEADS  32
#define HD     64
#define NFR    21
#define NA     32
#define ADIM   768
#define KVROWS (NFR * NA)     // 672
#define KVOUT  (2 * CDIM)     // 4096
#define SLEN   1560
#define TPF    98             // ceil(1560/16) token tiles per (frame, head)
#define LN1E4_OVER_64 0.14391157f   // ln(10000)/64

// ---------------------------------------------------------------------------
// small prep kernels
// ---------------------------------------------------------------------------
__global__ void cvt_bf16_kernel(const float* __restrict__ in,
                                unsigned short* __restrict__ out, int n) {
    int i = blockIdx.x * 256 + threadIdx.x;
    if (i < n) out[i] = f2bf(in[i]);
}

__global__ void minmax_kernel(const float* __restrict__ map, int n,
                              float* __restrict__ out4) {
    __shared__ float red[4][256];
    int tid = threadIdx.x;
    float mn0 = 3.4e38f, mx0 = -3.4e38f, mn1 = 3.4e38f, mx1 = -3.4e38f;
    for (int i = tid; i < n; i += 256) {
        float a = map[i], b = map[n + i];
        mn0 = fminf(mn0, a); mx0 = fmaxf(mx0, a);
        mn1 = fminf(mn1, b); mx1 = fmaxf(mx1, b);
    }
    red[0][tid] = mn0; red[1][tid] = mx0; red[2][tid] = mn1; red[3][tid] = mx1;
    __syncthreads();
    for (int s = 128; s > 0; s >>= 1) {
        if (tid < s) {
            red[0][tid] = fminf(red[0][tid], red[0][tid + s]);
            red[1][tid] = fmaxf(red[1][tid], red[1][tid + s]);
            red[2][tid] = fminf(red[2][tid], red[2][tid + s]);
            red[3][tid] = fmaxf(red[3][tid], red[3][tid + s]);
        }
        __syncthreads();
    }
    if (tid == 0) {
        out4[0] = red[0][0]; out4[1] = red[1][0];
        out4[2] = red[2][0]; out4[3] = red[3][0];
    }
}

__global__ void pos_kernel(const float* __restrict__ map,
                           const float* __restrict__ mm4,
                           float* __restrict__ pos, int n) {
    int i = blockIdx.x * 256 + threadIdx.x;
    if (i >= n) return;
    float r0 = map[i], r1 = map[n + i];
    float h1 = (r0 - mm4[0]) / (mm4[1] - mm4[0] + 1e-8f) * 4.0f;            // -> [0,4]
    float h2 = (r1 - mm4[2]) / (mm4[3] - mm4[2] + 1e-8f) * 4.0f + 20.0f;    // -> [20,24]
    pos[i] = (r0 >= r1) ? h1 : h2;   // argmax over the two rows (ties -> first)
}

// ---------------------------------------------------------------------------
// narrow GEMM loop (16x32 wave tile) -- used by the small KV GEMM.
// ---------------------------------------------------------------------------
template <int K, int LDB>
__device__ __forceinline__ void mm_loop(const unsigned short* __restrict__ arow,
                                        const unsigned short* __restrict__ bcol0,
                                        v8f& acc0, v8f& acc1) {
    const unsigned short* bcol1 = bcol0 + 16 * LDB;
#pragma unroll 4
    for (int k0 = 0; k0 < K; k0 += 32) {
        Frag a, b0, b1;
        a.q[0]  = *(const v4u*)(arow  + k0);
        a.q[1]  = *(const v4u*)(arow  + k0 + 16);
        b0.q[0] = *(const v4u*)(bcol0 + k0);
        b0.q[1] = *(const v4u*)(bcol0 + k0 + 16);
        b1.q[0] = *(const v4u*)(bcol1 + k0);
        b1.q[1] = *(const v4u*)(bcol1 + k0 + 16);
        __builtin_prefetch(arow + k0 + 256, 0, 1);
        acc0 = WMMA_BF16(a.v, b0.v, acc0);
        acc1 = WMMA_BF16(a.v, b1.v, acc1);
    }
}

// ---------------------------------------------------------------------------
// wide GEMM loop (32x64 wave tile): 2 A-frags + 4 B-frags -> 8 WMMAs per
// k-step. Doubles register-level reuse (~22 FLOP/B from L2) vs the narrow
// tile, which is what the WMMA pipes need given both operands are L2
// resident (192 MB L2 >> operand footprints).
// ---------------------------------------------------------------------------
template <int K, int LDB>
__device__ __forceinline__ void mm_loop_2x4(const unsigned short* __restrict__ a0,
                                            const unsigned short* __restrict__ a1,
                                            const unsigned short* __restrict__ b0,
                                            v8f acc[2][4]) {
#pragma unroll 2
    for (int k0 = 0; k0 < K; k0 += 32) {
        Frag A0, A1, B[4];
        A0.q[0] = *(const v4u*)(a0 + k0);
        A0.q[1] = *(const v4u*)(a0 + k0 + 16);
        A1.q[0] = *(const v4u*)(a1 + k0);
        A1.q[1] = *(const v4u*)(a1 + k0 + 16);
#pragma unroll
        for (int t = 0; t < 4; ++t) {
            const unsigned short* bc = b0 + t * 16 * LDB;
            B[t].q[0] = *(const v4u*)(bc + k0);
            B[t].q[1] = *(const v4u*)(bc + k0 + 16);
        }
        __builtin_prefetch(a0 + k0 + 256, 0, 1);
        __builtin_prefetch(a1 + k0 + 256, 0, 1);
#pragma unroll
        for (int t = 0; t < 4; ++t) {
            acc[0][t] = WMMA_BF16(A0.v, B[t].v, acc[0][t]);
            acc[1][t] = WMMA_BF16(A1.v, B[t].v, acc[1][t]);
        }
    }
}

// ---------------------------------------------------------------------------
// Q = x @ q_w^T + q_b, fused RoPE(normalized_pos) epilogue, bf16 output.
// Block tile 128x128, wave tile 32x64.
// ---------------------------------------------------------------------------
__global__ __launch_bounds__(256) void qgemm_rope_kernel(
        const unsigned short* __restrict__ Xb,
        const unsigned short* __restrict__ Wb,
        const float* __restrict__ bias,
        const float* __restrict__ pos,
        unsigned short* __restrict__ Qout, int M) {
    int lane = threadIdx.x & 31;
    int w    = threadIdx.x >> 5;
    int m0   = blockIdx.x * 128 + (w >> 1) * 32;
    int n0   = blockIdx.y * 128 + (w & 1) * 64;

    int mr0 = m0 + (lane & 15);      if (mr0 > M - 1) mr0 = M - 1;   // clamp
    int mr1 = m0 + 16 + (lane & 15); if (mr1 > M - 1) mr1 = M - 1;   // keep EXEC full
    const unsigned short* a0 = Xb + (size_t)mr0 * CDIM + ((lane < 16) ? 0 : 8);
    const unsigned short* a1 = Xb + (size_t)mr1 * CDIM + ((lane < 16) ? 0 : 8);
    const unsigned short* b0 = Wb + (size_t)(n0 + (lane & 15)) * CDIM + ((lane < 16) ? 0 : 8);

    v8f acc[2][4] = {};
    mm_loop_2x4<CDIM, CDIM>(a0, a1, b0, acc);

    int rbase = (lane >> 4) << 3;
#pragma unroll
    for (int i = 0; i < 2; ++i) {
#pragma unroll
        for (int t = 0; t < 4; ++t) {
            int col = n0 + t * 16 + (lane & 15);
            float b = bias[col];
            float twoj = (float)((col & 63) & ~1);
            float invf = __expf(-twoj * LN1E4_OVER_64);     // theta^-(2j/D)
#pragma unroll
            for (int e = 0; e < 8; ++e) {
                int row = m0 + i * 16 + rbase + e;
                int rc  = (row < M) ? row : (M - 1);
                float p = pos[rc];
                float f = p * invf;
                float c = __cosf(f), s = __sinf(f);
                float val = acc[i][t][e] + b;
                float partner = __shfl_xor(val, 1);         // pair lives in lane^1
                float res = val * c + (((col & 1) == 0) ? -partner : partner) * s;
                if (row < M) Qout[(size_t)row * CDIM + col] = f2bf(res);
            }
        }
    }
}

// ---------------------------------------------------------------------------
// KV = enc @ kv_w^T + kv_b. K half gets fixed-position RoPE and is stored
// [f, a, h, d]; V half is stored transposed [f, h, d, a].
// ---------------------------------------------------------------------------
__global__ __launch_bounds__(256) void kvgemm_kernel(
        const unsigned short* __restrict__ Eb,
        const unsigned short* __restrict__ Wb,
        const float* __restrict__ bias,
        unsigned short* __restrict__ Kout,
        unsigned short* __restrict__ Vout) {
    int lane = threadIdx.x & 31;
    int w    = threadIdx.x >> 5;
    int m0   = blockIdx.x * 64 + (w >> 1) * 16;
    int n0   = blockIdx.y * 64 + (w & 1) * 32;

    int mrow = m0 + (lane & 15);
    if (mrow > KVROWS - 1) mrow = KVROWS - 1;
    const unsigned short* arow  = Eb + (size_t)mrow * ADIM + ((lane < 16) ? 0 : 8);
    const unsigned short* bcol0 = Wb + (size_t)(n0 + (lane & 15)) * ADIM + ((lane < 16) ? 0 : 8);

    v8f acc0 = {}, acc1 = {};
    mm_loop<ADIM, ADIM>(arow, bcol0, acc0, acc1);

    int colbase = n0 + (lane & 15);
    int rbase   = (lane >> 4) << 3;
#pragma unroll
    for (int j = 0; j < 2; ++j) {
        v8f A = j ? acc1 : acc0;
        int col = colbase + j * 16;
        float b = bias[col];
        float twoj = (float)((col & 63) & ~1);
        float invf = __expf(-twoj * LN1E4_OVER_64);
        bool isK = (col < CDIM);
#pragma unroll
        for (int e = 0; e < 8; ++e) {
            int row = m0 + rbase + e;
            int a_local = row & 31;
            float val = A[e] + b;
            float pv = (a_local < 16) ? 2.0f : 22.0f;    // midpoints of H1/H2 ranges
            float f = pv * invf;
            float c = __cosf(f), s = __sinf(f);
            float partner = __shfl_xor(val, 1);          // executed by all lanes
            float roped = val * c + (((col & 1) == 0) ? -partner : partner) * s;
            if (row < KVROWS) {
                if (isK) {
                    Kout[(size_t)row * CDIM + col] = f2bf(roped);
                } else {
                    int o = col - CDIM;
                    int hh = o >> 6, dd = o & 63, fi = row >> 5;
                    Vout[(((size_t)fi * HEADS + hh) * HD + dd) * NA + a_local] = f2bf(val);
                }
            }
        }
    }
}

// ---------------------------------------------------------------------------
// Attention: one wave per (frame, head, 16-token tile).
// scores (16x32) via WMMA, in-register softmax over N_a=32, P staged through
// LDS to become an A-fragment, then P @ V (16x64) via WMMA.
// ---------------------------------------------------------------------------
__global__ __launch_bounds__(256) void attn_kernel(
        const unsigned short* __restrict__ Qb,
        const unsigned short* __restrict__ Kb,
        const unsigned short* __restrict__ Vb,
        unsigned short* __restrict__ Ob) {
    __shared__ alignas(16) unsigned short pshm[8][512];   // 16x32 bf16 per wave

    int lane = threadIdx.x & 31;
    int w    = threadIdx.x >> 5;
    int gw   = blockIdx.x * 8 + w;
    const int total = NFR * HEADS * TPF;
    if (gw >= total) return;                               // wave-uniform exit

    int f   = gw / (HEADS * TPF);
    int rem = gw - f * (HEADS * TPF);
    int h   = rem / TPF;
    int t   = rem - h * TPF;
    int sbase = t * 16;

    int srow = sbase + (lane & 15);
    if (srow > SLEN - 1) srow = SLEN - 1;
    const unsigned short* qrow = Qb + ((size_t)(f * SLEN + srow)) * CDIM + h * HD
                                    + ((lane < 16) ? 0 : 8);
    const unsigned short* kcol = Kb + ((size_t)(f * NA + (lane & 15))) * CDIM + h * HD
                                    + ((lane < 16) ? 0 : 8);

    // scores: Q (16x64) @ K^T (64x32) -> two 16x16 accumulators, K-dim = 2 steps
    v8f sA = {}, sB = {};
#pragma unroll
    for (int k0 = 0; k0 < HD; k0 += 32) {
        Frag a, b0, b1;
        a.q[0]  = *(const v4u*)(qrow + k0);
        a.q[1]  = *(const v4u*)(qrow + k0 + 16);
        b0.q[0] = *(const v4u*)(kcol + k0);
        b0.q[1] = *(const v4u*)(kcol + k0 + 16);
        b1.q[0] = *(const v4u*)(kcol + 16 * CDIM + k0);
        b1.q[1] = *(const v4u*)(kcol + 16 * CDIM + k0 + 16);
        sA = WMMA_BF16(a.v, b0.v, sA);
        sB = WMMA_BF16(a.v, b1.v, sB);
    }

    // softmax over the 32 columns of each row. Row values live across the 16
    // lanes of each half-wave, so xor-reduce with offsets 1,2,4,8 only.
    const float scale = 0.125f;   // 1/sqrt(64)
#pragma unroll
    for (int e = 0; e < 8; ++e) {
        float m = fmaxf(sA[e], sB[e]) * scale;
#pragma unroll
        for (int off = 1; off < 16; off <<= 1) m = fmaxf(m, __shfl_xor(m, off));
        float eA = __expf(sA[e] * scale - m);
        float eB = __expf(sB[e] * scale - m);
        float sum = eA + eB;
#pragma unroll
        for (int off = 1; off < 16; off <<= 1) sum += __shfl_xor(sum, off);
        float inv = 1.0f / sum;
        sA[e] = eA * inv;
        sB[e] = eB * inv;
    }

    // stage P (16x32) to LDS row-major, then reload as an A-fragment
    int rbase = (lane >> 4) << 3;
#pragma unroll
    for (int e = 0; e < 8; ++e) {
        pshm[w][(rbase + e) * 32 + (lane & 15)]      = f2bf(sA[e]);
        pshm[w][(rbase + e) * 32 + 16 + (lane & 15)] = f2bf(sB[e]);
    }
    asm volatile("s_wait_dscnt 0x0" ::: "memory");   // wave-local LDS RAW fence

    Frag pa;
    const unsigned short* pp = &pshm[w][(lane & 15) * 32 + ((lane < 16) ? 0 : 8)];
    pa.q[0] = *(const v4u*)(pp);
    pa.q[1] = *(const v4u*)(pp + 16);

    // out (16x64) = P (16x32) @ V (32x64); V stored [f,h,d,a] so columns are
    // contiguous along K=a.
    const unsigned short* vb = Vb + ((size_t)(f * HEADS + h)) * HD * NA;
    v8f o[4];
#pragma unroll
    for (int j = 0; j < 4; ++j) {
        Frag b;
        const unsigned short* vc = vb + (j * 16 + (lane & 15)) * NA + ((lane < 16) ? 0 : 8);
        b.q[0] = *(const v4u*)(vc);
        b.q[1] = *(const v4u*)(vc + 16);
        v8f z = {};
        o[j] = WMMA_BF16(pa.v, b.v, z);
    }
#pragma unroll
    for (int j = 0; j < 4; ++j) {
#pragma unroll
        for (int e = 0; e < 8; ++e) {
            int s = sbase + rbase + e;
            if (s < SLEN)
                Ob[((size_t)(f * SLEN + s)) * CDIM + h * HD + j * 16 + (lane & 15)] =
                    f2bf(o[j][e]);
        }
    }
}

// ---------------------------------------------------------------------------
// out = attn_out @ proj_w^T + proj_b  (fp32 output).
// Block tile 128x128, wave tile 32x64.
// ---------------------------------------------------------------------------
__global__ __launch_bounds__(256) void projgemm_kernel(
        const unsigned short* __restrict__ Ab,
        const unsigned short* __restrict__ Wb,
        const float* __restrict__ bias,
        float* __restrict__ Out, int M) {
    int lane = threadIdx.x & 31;
    int w    = threadIdx.x >> 5;
    int m0   = blockIdx.x * 128 + (w >> 1) * 32;
    int n0   = blockIdx.y * 128 + (w & 1) * 64;

    int mr0 = m0 + (lane & 15);      if (mr0 > M - 1) mr0 = M - 1;
    int mr1 = m0 + 16 + (lane & 15); if (mr1 > M - 1) mr1 = M - 1;
    const unsigned short* a0 = Ab + (size_t)mr0 * CDIM + ((lane < 16) ? 0 : 8);
    const unsigned short* a1 = Ab + (size_t)mr1 * CDIM + ((lane < 16) ? 0 : 8);
    const unsigned short* b0 = Wb + (size_t)(n0 + (lane & 15)) * CDIM + ((lane < 16) ? 0 : 8);

    v8f acc[2][4] = {};
    mm_loop_2x4<CDIM, CDIM>(a0, a1, b0, acc);

    int rbase = (lane >> 4) << 3;
#pragma unroll
    for (int i = 0; i < 2; ++i) {
#pragma unroll
        for (int t = 0; t < 4; ++t) {
            int col = n0 + t * 16 + (lane & 15);
            float b = bias[col];
#pragma unroll
            for (int e = 0; e < 8; ++e) {
                int row = m0 + i * 16 + rbase + e;
                if (row < M) Out[(size_t)row * CDIM + col] = acc[i][t][e] + b;
            }
        }
    }
}

// ---------------------------------------------------------------------------
// host side
// ---------------------------------------------------------------------------
extern "C" void kernel_launch(void* const* d_in, const int* in_sizes, int n_in,
                              void* d_out, int out_size, void* d_ws, size_t ws_size,
                              hipStream_t stream) {
    const float* x      = (const float*)d_in[0];
    const float* enc    = (const float*)d_in[1];
    const float* amap   = (const float*)d_in[2];
    const float* q_w    = (const float*)d_in[3];
    const float* q_b    = (const float*)d_in[4];
    const float* kv_w   = (const float*)d_in[5];
    const float* kv_b   = (const float*)d_in[6];
    const float* proj_w = (const float*)d_in[7];
    const float* proj_b = (const float*)d_in[8];
    (void)in_sizes; (void)n_in; (void)out_size; (void)ws_size;

    char* ws = (char*)d_ws;
    size_t off = 0;
    auto take = [&](size_t bytes) {
        size_t r = off;
        off = (off + bytes + 255) & ~(size_t)255;
        return r;
    };

    float*          mm4  = (float*)(ws + take(4 * sizeof(float)));
    float*          pos  = (float*)(ws + take((size_t)NTOK * sizeof(float)));
    unsigned short* xb   = (unsigned short*)(ws + take((size_t)NTOK * CDIM * 2));
    unsigned short* encb = (unsigned short*)(ws + take((size_t)KVROWS * ADIM * 2));
    unsigned short* qwb  = (unsigned short*)(ws + take((size_t)CDIM * CDIM * 2));
    unsigned short* kvwb = (unsigned short*)(ws + take((size_t)KVOUT * ADIM * 2));
    unsigned short* pwb  = (unsigned short*)(ws + take((size_t)CDIM * CDIM * 2));
    unsigned short* qb16 = (unsigned short*)(ws + take((size_t)NTOK * CDIM * 2));
    unsigned short* kb16 = (unsigned short*)(ws + take((size_t)KVROWS * CDIM * 2));
    unsigned short* vb16 = (unsigned short*)(ws + take((size_t)NFR * HEADS * HD * NA * 2));
    unsigned short* ob16 = (unsigned short*)(ws + take((size_t)NTOK * CDIM * 2));

    // precision conversions (bf16 feeds the WMMA pipes)
    {
        int n;
        n = NTOK * CDIM;    cvt_bf16_kernel<<<(n + 255) / 256, 256, 0, stream>>>(x, xb, n);
        n = KVROWS * ADIM;  cvt_bf16_kernel<<<(n + 255) / 256, 256, 0, stream>>>(enc, encb, n);
        n = CDIM * CDIM;    cvt_bf16_kernel<<<(n + 255) / 256, 256, 0, stream>>>(q_w, qwb, n);
        n = KVOUT * ADIM;   cvt_bf16_kernel<<<(n + 255) / 256, 256, 0, stream>>>(kv_w, kvwb, n);
        n = CDIM * CDIM;    cvt_bf16_kernel<<<(n + 255) / 256, 256, 0, stream>>>(proj_w, pwb, n);
    }

    // RoPE positions from x_ref_attn_map
    minmax_kernel<<<1, 256, 0, stream>>>(amap, NTOK, mm4);
    pos_kernel<<<(NTOK + 255) / 256, 256, 0, stream>>>(amap, mm4, pos, NTOK);

    // Q projection + RoPE   (grid: 256 x 16 blocks of 128x128)
    qgemm_rope_kernel<<<dim3((NTOK + 127) / 128, CDIM / 128), 256, 0, stream>>>(
        xb, qwb, q_b, pos, qb16, NTOK);

    // KV projection (+K RoPE, V transpose)
    kvgemm_kernel<<<dim3((KVROWS + 63) / 64, KVOUT / 64), 256, 0, stream>>>(
        encb, kvwb, kv_b, kb16, vb16);

    // attention
    {
        int waves  = NFR * HEADS * TPF;        // 65856
        int blocks = (waves + 7) / 8;          // 8232
        attn_kernel<<<blocks, 256, 0, stream>>>(qb16, kb16, vb16, ob16);
    }

    // output projection
    projgemm_kernel<<<dim3((NTOK + 127) / 128, CDIM / 128), 256, 0, stream>>>(
        ob16, pwb, proj_b, (float*)d_out, NTOK);
}
